// Block_22823456211331
// MI455X (gfx1250) — matrix operand order repaired and checked by reading, not compiled, not run on hardware
//
#include <hip/hip_runtime.h>

// ---------------------------------------------------------------------------
// Transformer block for MI455X (gfx1250, wave32).
// GEMMs: v_wmma_f32_16x16x32_bf16, ping-pong double-buffered fragments,
// all loads immediate-offset from running base pointers.  Tile sizes chosen
// to stay under the 256-VGPR budget (no scratch spills).
// Attention: flash-style online softmax.
// ---------------------------------------------------------------------------

typedef __bf16 bf16_t;
typedef __bf16 v8bf  __attribute__((ext_vector_type(8)));
typedef __bf16 v16bf __attribute__((ext_vector_type(16)));
typedef float  v8f   __attribute__((ext_vector_type(8)));

#define B_  2
#define S_  2048
#define E_  1024
#define H_  16
#define HS_ 64
#define BS_ (B_ * S_)

__device__ __forceinline__ v8f vzero8() {
  v8f z = {0.f, 0.f, 0.f, 0.f, 0.f, 0.f, 0.f, 0.f};
  return z;
}

__device__ __forceinline__ v16bf cat16(v8bf lo, v8bf hi) {
  v16bf r;
#pragma unroll
  for (int i = 0; i < 8; ++i) { r[i] = lo[i]; r[i + 8] = hi[i]; }
  return r;
}

__device__ __forceinline__ v8f wmma_bf16(v16bf a, v16bf b, v8f c) {
  return __builtin_amdgcn_wmma_f32_16x16x32_bf16(false, a, false, b, (short)0, c,
                                                 false, false);
}

// A fragment given a lane-resolved base (row + half*8 already applied).
__device__ __forceinline__ v16bf frag_a(const bf16_t* p) {
  return cat16(*(const v8bf*)p, *(const v8bf*)(p + 16));
}
// B fragment given a lane-resolved base (row n + half*8 already applied).
__device__ __forceinline__ v16bf frag_b(const bf16_t* p) {
  return cat16(*(const v8bf*)p, *(const v8bf*)(p + 16));
}

// Generic fragment loaders used by attention (address per call).
__device__ __forceinline__ v16bf load_a(const bf16_t* Abase, int lda, int kk, int lane) {
  const int half = (lane >> 4) & 1;
  return frag_a(Abase + (size_t)(lane & 15) * lda + kk + half * 8);
}
__device__ __forceinline__ v16bf load_b(const bf16_t* Bt, size_t ldk, int n, int kk,
                                        int lane) {
  const int half = (lane >> 4) & 1;
  return frag_b(Bt + (size_t)n * ldk + kk + half * 8);
}

// ---------------------------------------------------------------------------
// Tiled transpose: fp32 [K,N] -> bf16 [N,K], 32x32 tiles through LDS.
// ---------------------------------------------------------------------------
__global__ void k_transpose_bf16(const float* __restrict__ W, bf16_t* __restrict__ Wt,
                                 int K, int N) {
  __shared__ float tile[32][33];
  const int k0 = blockIdx.x * 32;
  const int n0 = blockIdx.y * 32;
  const int tx = threadIdx.x & 31;
  const int ty = threadIdx.x >> 5;
#pragma unroll
  for (int i = 0; i < 4; ++i) {
    int k = ty + i * 8;
    tile[k][tx] = W[(size_t)(k0 + k) * N + n0 + tx];
  }
  __syncthreads();
#pragma unroll
  for (int i = 0; i < 4; ++i) {
    int n = ty + i * 8;
    Wt[(size_t)(n0 + n) * K + k0 + tx] = (bf16_t)tile[tx][n];
  }
}

// Wq/Wk/Wv: fp32 (H, E, HS) -> bf16 [n = h*HS + d, k = e], tiled per head.
__global__ void k_conv_wqkv(const float* __restrict__ W, bf16_t* __restrict__ Wt) {
  __shared__ float tile[32][33];
  const int h = blockIdx.z;
  const int k0 = blockIdx.x * 32;  // e
  const int d0 = blockIdx.y * 32;  // d
  const int tx = threadIdx.x & 31;
  const int ty = threadIdx.x >> 5;
#pragma unroll
  for (int i = 0; i < 4; ++i) {
    int e = ty + i * 8;
    tile[e][tx] = W[((size_t)h * E_ + k0 + e) * HS_ + d0 + tx];
  }
  __syncthreads();
#pragma unroll
  for (int i = 0; i < 4; ++i) {
    int d = ty + i * 8;
    Wt[((size_t)h * HS_ + d0 + d) * E_ + k0 + tx] = (bf16_t)tile[tx][d];
  }
}

// ---------------------------------------------------------------------------
// RMSNorm: fp32 row -> bf16 row.  One block (256 thr) per row of E=1024.
// ---------------------------------------------------------------------------
__global__ void k_rmsnorm(const float* __restrict__ x, const float* __restrict__ w,
                          bf16_t* __restrict__ y) {
  const int row = blockIdx.x;
  const float* xr = x + (size_t)row * E_;
  float ss = 0.f;
  for (int i = threadIdx.x; i < E_; i += 256) { float v = xr[i]; ss += v * v; }
#pragma unroll
  for (int m = 16; m >= 1; m >>= 1) ss += __shfl_xor(ss, m, 32);
  __shared__ float red[8];
  if ((threadIdx.x & 31) == 0) red[threadIdx.x >> 5] = ss;
  __syncthreads();
  float tot = 0.f;
#pragma unroll
  for (int i = 0; i < 8; ++i) tot += red[i];
  const float r = rsqrtf(tot * (1.0f / E_) + 1e-6f);
  bf16_t* yr = y + (size_t)row * E_;
  for (int i = threadIdx.x; i < E_; i += 256) yr[i] = (bf16_t)(xr[i] * r * w[i]);
}

// ---------------------------------------------------------------------------
// RoPE (in place, bf16 buffer [b,h,s,d]); optional scale folded into q.
// ---------------------------------------------------------------------------
__global__ void k_rope(bf16_t* __restrict__ qk, const float* __restrict__ cs,
                       const float* __restrict__ sn, float scale) {
  int idx = blockIdx.x * blockDim.x + threadIdx.x;
  if (idx >= B_ * H_ * S_ * (HS_ / 2)) return;
  int j = idx & 31;
  int s = (idx >> 5) & (S_ - 1);
  float c = cs[s * 32 + j], si = sn[s * 32 + j];
  bf16_t* p = qk + ((size_t)idx << 1);
  float xr = (float)p[0], xi = (float)p[1];
  p[0] = (bf16_t)((xr * c - xi * si) * scale);
  p[1] = (bf16_t)((xr * si + xi * c) * scale);
}

// ---------------------------------------------------------------------------
// QKV GEMM: h1[BS,E] @ Wt[N=E,K=E]; 32x64 tile per wave, ping-pong buffered.
// OUTMODE 0: write bf16 [b,h,s,d]; OUTMODE 1: write bf16 vT [b,h,d,s].
// ---------------------------------------------------------------------------
template <int OUTMODE>
__global__ void k_gemm_qkv(const bf16_t* __restrict__ A, const bf16_t* __restrict__ Wt,
                           bf16_t* __restrict__ out) {
  const int lane = threadIdx.x;
  const int nl = lane & 15;
  const int half = (lane >> 4) & 1;
  const int m0 = blockIdx.x * 32;
  const int n0 = blockIdx.y * 64;
  const int GA = 16 * E_;
  const int GB = 16 * E_;

  const bf16_t* pa = A + ((size_t)m0 + nl) * E_ + half * 8;
  const bf16_t* pb = Wt + (size_t)(n0 + nl) * E_ + half * 8;

  v8f acc[2][4];
#pragma unroll
  for (int t = 0; t < 2; ++t)
#pragma unroll
    for (int c = 0; c < 4; ++c) acc[t][c] = vzero8();

  v16bf aA0, aA1, bA[4], aB0, aB1, bB[4];

  auto LDA = [&](v16bf& x0, v16bf& x1, v16bf* bx, int off) {
    x0 = frag_a(pa + off);
    x1 = frag_a(pa + GA + off);
#pragma unroll
    for (int c = 0; c < 4; ++c) bx[c] = frag_b(pb + c * GB + off);
  };
  auto MM = [&](v16bf& x0, v16bf& x1, v16bf* bx) {
#pragma unroll
    for (int c = 0; c < 4; ++c) {
      acc[0][c] = wmma_bf16(x0, bx[c], acc[0][c]);
      acc[1][c] = wmma_bf16(x1, bx[c], acc[1][c]);
    }
  };

  LDA(aA0, aA1, bA, 0);
  for (int it = 0; it < E_ / 64 - 1; ++it) {
    LDA(aB0, aB1, bB, 32);
    MM(aA0, aA1, bA);
    pa += 64;
    pb += 64;
    LDA(aA0, aA1, bA, 0);
    MM(aB0, aB1, bB);
  }
  LDA(aB0, aB1, bB, 32);
  MM(aA0, aA1, bA);
  MM(aB0, aB1, bB);

  const int rbase = half * 8;
  const int bb_ = m0 >> 11;  // row / S
#pragma unroll
  for (int t = 0; t < 2; ++t) {
    const int s0 = (m0 & (S_ - 1)) + t * 16;
#pragma unroll
    for (int c = 0; c < 4; ++c) {
      const int col = n0 + c * 16 + nl;
      const int h = col >> 6, d = col & 63;
      if (OUTMODE == 0) {
#pragma unroll
        for (int r = 0; r < 8; ++r) {
          size_t ad = (((size_t)(bb_ * H_ + h) * S_) + s0 + rbase + r) * HS_ + d;
          out[ad] = (bf16_t)acc[t][c][r];
        }
      } else {
        v8bf st;
#pragma unroll
        for (int r = 0; r < 8; ++r) st[r] = (bf16_t)acc[t][c][r];
        size_t ad = ((size_t)(bb_ * H_ + h) * HS_ + d) * S_ + s0 + rbase;
        *(v8bf*)(out + ad) = st;
      }
    }
  }
}

// ---------------------------------------------------------------------------
// Residual GEMM: out_f32 = A[BS,K] @ Wt[1024,K] + resid + bias.
// 32x64 tile per wave, ping-pong buffered.
// ---------------------------------------------------------------------------
template <int KTOT>
__global__ void k_gemm_res(const bf16_t* __restrict__ A, const bf16_t* __restrict__ Wt,
                           const float* __restrict__ resid, const float* __restrict__ bias,
                           float* __restrict__ out) {
  const int lane = threadIdx.x;
  const int nl = lane & 15;
  const int half = (lane >> 4) & 1;
  const int m0 = blockIdx.x * 32;
  const int n0 = blockIdx.y * 64;
  const int GA = 16 * KTOT;
  const int GB = 16 * KTOT;

  const bf16_t* pa = A + ((size_t)m0 + nl) * KTOT + half * 8;
  const bf16_t* pb = Wt + (size_t)(n0 + nl) * KTOT + half * 8;

  v8f acc[2][4];
#pragma unroll
  for (int t = 0; t < 2; ++t)
#pragma unroll
    for (int c = 0; c < 4; ++c) acc[t][c] = vzero8();

  v16bf aA0, aA1, bA[4], aB0, aB1, bB[4];

  auto LDA = [&](v16bf& x0, v16bf& x1, v16bf* bx, int off) {
    x0 = frag_a(pa + off);
    x1 = frag_a(pa + GA + off);
#pragma unroll
    for (int c = 0; c < 4; ++c) bx[c] = frag_b(pb + c * GB + off);
  };
  auto MM = [&](v16bf& x0, v16bf& x1, v16bf* bx) {
#pragma unroll
    for (int c = 0; c < 4; ++c) {
      acc[0][c] = wmma_bf16(x0, bx[c], acc[0][c]);
      acc[1][c] = wmma_bf16(x1, bx[c], acc[1][c]);
    }
  };

  LDA(aA0, aA1, bA, 0);
  for (int it = 0; it < KTOT / 64 - 1; ++it) {
    LDA(aB0, aB1, bB, 32);
    MM(aA0, aA1, bA);
    pa += 64;
    pb += 64;
    LDA(aA0, aA1, bA, 0);
    MM(aB0, aB1, bB);
  }
  LDA(aB0, aB1, bB, 32);
  MM(aA0, aA1, bA);
  MM(aB0, aB1, bB);

  const int rbase = half * 8;
#pragma unroll
  for (int t = 0; t < 2; ++t) {
#pragma unroll
    for (int c = 0; c < 4; ++c) {
      const int col = n0 + c * 16 + nl;
      const float bv = bias[col];
#pragma unroll
      for (int r = 0; r < 8; ++r) {
        size_t ad = (size_t)(m0 + t * 16 + rbase + r) * E_ + col;
        out[ad] = acc[t][c][r] + resid[ad] + bv;
      }
    }
  }
}

// ---------------------------------------------------------------------------
// Fused SwiGLU GEMM: u = h2 @ W1 + b1; f = u[:,:4E] * sigmoid(u[:,4E:]).
// 32 rows x 32 cols of BOTH halves per wave (8 accum v8f + 2x6 frag v16bf
// ~= 190 VGPRs: fits the 256-VGPR budget, no scratch spills).
// ---------------------------------------------------------------------------
__global__ void k_gemm_swiglu(const bf16_t* __restrict__ A, const bf16_t* __restrict__ W1t,
                              const float* __restrict__ b1, bf16_t* __restrict__ f) {
  const int lane = threadIdx.x;
  const int nl = lane & 15;
  const int half = (lane >> 4) & 1;
  const int m0 = blockIdx.x * 32;
  const int n0 = blockIdx.y * 32;
  const int GA = 16 * E_;
  const int GB = 16 * E_;

  const bf16_t* pa = A + ((size_t)m0 + nl) * E_ + half * 8;
  const bf16_t* pb = W1t + (size_t)(n0 + nl) * E_ + half * 8;
  const bf16_t* pg = pb + (size_t)4096 * E_;

  v8f aa[2][2], ag[2][2];
#pragma unroll
  for (int t = 0; t < 2; ++t)
#pragma unroll
    for (int c = 0; c < 2; ++c) { aa[t][c] = vzero8(); ag[t][c] = vzero8(); }

  v16bf aA0, aA1, bA[2], gA[2], aB0, aB1, bB[2], gB[2];

  auto LDA = [&](v16bf& x0, v16bf& x1, v16bf* bx, v16bf* gx, int off) {
    x0 = frag_a(pa + off);
    x1 = frag_a(pa + GA + off);
#pragma unroll
    for (int c = 0; c < 2; ++c) {
      bx[c] = frag_b(pb + c * GB + off);
      gx[c] = frag_b(pg + c * GB + off);
    }
  };
  auto MM = [&](v16bf& x0, v16bf& x1, v16bf* bx, v16bf* gx) {
#pragma unroll
    for (int c = 0; c < 2; ++c) {
      aa[0][c] = wmma_bf16(x0, bx[c], aa[0][c]);
      aa[1][c] = wmma_bf16(x1, bx[c], aa[1][c]);
      ag[0][c] = wmma_bf16(x0, gx[c], ag[0][c]);
      ag[1][c] = wmma_bf16(x1, gx[c], ag[1][c]);
    }
  };

  LDA(aA0, aA1, bA, gA, 0);
  for (int it = 0; it < E_ / 64 - 1; ++it) {
    LDA(aB0, aB1, bB, gB, 32);
    MM(aA0, aA1, bA, gA);
    pa += 64;
    pb += 64;
    pg += 64;
    LDA(aA0, aA1, bA, gA, 0);
    MM(aB0, aB1, bB, gB);
  }
  LDA(aB0, aB1, bB, gB, 32);
  MM(aA0, aA1, bA, gA);
  MM(aB0, aB1, bB, gB);

  const int rbase = half * 8;
#pragma unroll
  for (int t = 0; t < 2; ++t) {
#pragma unroll
    for (int c = 0; c < 2; ++c) {
      const int col = n0 + c * 16 + nl;
      const float bav = b1[col], bgv = b1[col + 4096];
#pragma unroll
      for (int r = 0; r < 8; ++r) {
        float av = aa[t][c][r] + bav;
        float gv = ag[t][c][r] + bgv;
        float sv = 1.0f / (1.0f + __expf(-gv));
        f[(size_t)(m0 + t * 16 + rbase + r) * 4096 + col] = (bf16_t)(av * sv);
      }
    }
  }
}

// ---------------------------------------------------------------------------
// Flash attention. One wave per (b, h, 16-query tile).
// q,k: bf16 [b,h,s,d] (q pre-scaled by HS^-0.5); vT: bf16 [b,h,d,s].
// o: bf16 [b, s, h*64+d]  (layout ready for the Wo GEMM).
// ---------------------------------------------------------------------------
__global__ void k_attn(const bf16_t* __restrict__ q, const bf16_t* __restrict__ k,
                       const bf16_t* __restrict__ vT, bf16_t* __restrict__ o) {
  __shared__ __align__(16) bf16_t pbuf[16 * 32];
  const int lane = threadIdx.x;
  const int bid = blockIdx.x;
  const int qt = bid & 127;
  const int bh = bid >> 7;
  const int s0 = qt * 16;
  const int nl = lane & 15;
  const int half = (lane >> 4) & 1;
  const int rbase = half * 8;

  const bf16_t* qbase = q + ((size_t)bh * S_ + s0) * HS_;
  const bf16_t* kbase = k + (size_t)bh * S_ * HS_;
  const bf16_t* vbase = vT + (size_t)bh * HS_ * S_;

  const v16bf qa0 = load_a(qbase, HS_, 0, lane);
  const v16bf qa1 = load_a(qbase, HS_, 32, lane);

  v8f oacc[4] = {vzero8(), vzero8(), vzero8(), vzero8()};
  float mrow[8], lrow[8];
#pragma unroll
  for (int r = 0; r < 8; ++r) { mrow[r] = -1e30f; lrow[r] = 0.f; }

  for (int t0 = 0; t0 < s0 + 16; t0 += 32) {
    v8f sc0 = vzero8(), sc1 = vzero8();
    const bf16_t* kt = kbase + (size_t)t0 * HS_;
    sc0 = wmma_bf16(qa0, load_b(kt, HS_, nl, 0, lane), sc0);
    sc0 = wmma_bf16(qa1, load_b(kt, HS_, nl, 32, lane), sc0);
    sc1 = wmma_bf16(qa0, load_b(kt, HS_, 16 + nl, 0, lane), sc1);
    sc1 = wmma_bf16(qa1, load_b(kt, HS_, 16 + nl, 32, lane), sc1);

#pragma unroll
    for (int r = 0; r < 8; ++r) {
      const int srow = s0 + rbase + r;
      float x0 = (t0 + nl > srow) ? -1e30f : sc0[r];
      float x1 = (t0 + 16 + nl > srow) ? -1e30f : sc1[r];
      float tm = fmaxf(x0, x1);
      tm = fmaxf(tm, __shfl_xor(tm, 1, 32));
      tm = fmaxf(tm, __shfl_xor(tm, 2, 32));
      tm = fmaxf(tm, __shfl_xor(tm, 4, 32));
      tm = fmaxf(tm, __shfl_xor(tm, 8, 32));
      const float mn = fmaxf(mrow[r], tm);
      const float alpha = __expf(mrow[r] - mn);
      const float p0 = __expf(x0 - mn);
      const float p1 = __expf(x1 - mn);
      float rs = p0 + p1;
      rs += __shfl_xor(rs, 1, 32);
      rs += __shfl_xor(rs, 2, 32);
      rs += __shfl_xor(rs, 4, 32);
      rs += __shfl_xor(rs, 8, 32);
      lrow[r] = lrow[r] * alpha + rs;
      mrow[r] = mn;
      oacc[0][r] *= alpha;
      oacc[1][r] *= alpha;
      oacc[2][r] *= alpha;
      oacc[3][r] *= alpha;
      pbuf[(rbase + r) * 32 + nl] = (bf16_t)p0;
      pbuf[(rbase + r) * 32 + 16 + nl] = (bf16_t)p1;
    }
    asm volatile("s_wait_dscnt 0" ::: "memory");

    const int k0 = half * 8;
    v8bf plo = *(const v8bf*)(&pbuf[nl * 32 + k0]);
    v8bf phi = *(const v8bf*)(&pbuf[nl * 32 + k0 + 16]);
    v16bf pa = cat16(plo, phi);

#pragma unroll
    for (int c = 0; c < 4; ++c) {
      v16bf vb = load_b(vbase, S_, c * 16 + nl, t0, lane);
      oacc[c] = wmma_bf16(pa, vb, oacc[c]);
    }
  }

  const int bb = bh >> 4, h = bh & 15;
#pragma unroll
  for (int r = 0; r < 8; ++r) {
    const float inv = 1.0f / lrow[r];
    const size_t base = ((size_t)bb * S_ + s0 + rbase + r) * E_ + (size_t)h * HS_;
#pragma unroll
    for (int c = 0; c < 4; ++c) o[base + c * 16 + nl] = (bf16_t)(oacc[c][r] * inv);
  }
}

// ---------------------------------------------------------------------------
// Host launch
// ---------------------------------------------------------------------------
extern "C" void kernel_launch(void* const* d_in, const int* in_sizes, int n_in,
                              void* d_out, int out_size, void* d_ws, size_t ws_size,
                              hipStream_t stream) {
  const float* x     = (const float*)d_in[0];
  const float* cosb  = (const float*)d_in[1];
  const float* sinb  = (const float*)d_in[2];
  const float* w_ln1 = (const float*)d_in[3];
  const float* Wq    = (const float*)d_in[4];
  const float* Wk    = (const float*)d_in[5];
  const float* Wv    = (const float*)d_in[6];
  const float* Wo    = (const float*)d_in[7];
  const float* bo    = (const float*)d_in[8];
  const float* w_ln2 = (const float*)d_in[9];
  const float* W1    = (const float*)d_in[10];
  const float* b1    = (const float*)d_in[11];
  const float* W2    = (const float*)d_in[12];
  const float* b2    = (const float*)d_in[13];
  float* out = (float*)d_out;

  char* ws = (char*)d_ws;
  size_t off = 0;
  auto alloc = [&](size_t bytes) -> void* {
    void* p = ws + off;
    off += (bytes + 255) & ~(size_t)255;
    return p;
  };
  bf16_t* h1  = (bf16_t*)alloc((size_t)BS_ * E_ * 2);
  bf16_t* qb  = (bf16_t*)alloc((size_t)BS_ * E_ * 2);
  bf16_t* kb  = (bf16_t*)alloc((size_t)BS_ * E_ * 2);
  bf16_t* vTb = (bf16_t*)alloc((size_t)BS_ * E_ * 2);
  bf16_t* ob  = (bf16_t*)alloc((size_t)BS_ * E_ * 2);
  float*  x1  = (float*)alloc((size_t)BS_ * E_ * 4);
  bf16_t* h2  = (bf16_t*)alloc((size_t)BS_ * E_ * 2);
  bf16_t* fb  = (bf16_t*)alloc((size_t)BS_ * 4096 * 2);
  bf16_t* Wqt = (bf16_t*)alloc((size_t)E_ * E_ * 2);
  bf16_t* Wkt = (bf16_t*)alloc((size_t)E_ * E_ * 2);
  bf16_t* Wvt = (bf16_t*)alloc((size_t)E_ * E_ * 2);
  bf16_t* Wot = (bf16_t*)alloc((size_t)E_ * E_ * 2);
  bf16_t* W1t = (bf16_t*)alloc((size_t)E_ * 8192 * 2);
  bf16_t* W2t = (bf16_t*)alloc((size_t)4096 * E_ * 2);

  // weight conversion (fp32 -> bf16, transposed to [N,K], LDS-tiled)
  k_conv_wqkv<<<dim3(E_ / 32, HS_ / 32, H_), 256, 0, stream>>>(Wq, Wqt);
  k_conv_wqkv<<<dim3(E_ / 32, HS_ / 32, H_), 256, 0, stream>>>(Wk, Wkt);
  k_conv_wqkv<<<dim3(E_ / 32, HS_ / 32, H_), 256, 0, stream>>>(Wv, Wvt);
  k_transpose_bf16<<<dim3(E_ / 32, E_ / 32), 256, 0, stream>>>(Wo, Wot, E_, E_);
  k_transpose_bf16<<<dim3(E_ / 32, 8192 / 32), 256, 0, stream>>>(W1, W1t, E_, 8192);
  k_transpose_bf16<<<dim3(4096 / 32, E_ / 32), 256, 0, stream>>>(W2, W2t, 4096, E_);

  // h1 = rmsnorm(x)
  k_rmsnorm<<<BS_, 256, 0, stream>>>(x, w_ln1, h1);

  // q, k (row layout), v (transposed layout)
  dim3 gq(BS_ / 32, E_ / 64);
  k_gemm_qkv<0><<<gq, 32, 0, stream>>>(h1, Wqt, qb);
  k_gemm_qkv<0><<<gq, 32, 0, stream>>>(h1, Wkt, kb);
  k_gemm_qkv<1><<<gq, 32, 0, stream>>>(h1, Wvt, vTb);

  // RoPE (scale HS^-0.5 folded into q)
  const int nrope = B_ * H_ * S_ * (HS_ / 2);
  k_rope<<<(nrope + 255) / 256, 256, 0, stream>>>(qb, cosb, sinb, 0.125f);
  k_rope<<<(nrope + 255) / 256, 256, 0, stream>>>(kb, cosb, sinb, 1.0f);

  // flash attention -> o [b, s, h*64+d]
  k_attn<<<B_ * H_ * (S_ / 16), 32, 0, stream>>>(qb, kb, vTb, ob);

  // x1 = x + o @ Wo + bo
  k_gemm_res<E_><<<dim3(BS_ / 32, E_ / 64), 32, 0, stream>>>(ob, Wot, x, bo, x1);

  // h2 = rmsnorm(x1)
  k_rmsnorm<<<BS_, 256, 0, stream>>>(x1, w_ln2, h2);

  // f = swiglu(h2 @ W1 + b1)
  k_gemm_swiglu<<<dim3(BS_ / 32, 4096 / 32), 32, 0, stream>>>(h2, W1t, b1, fb);

  // out = x1 + f @ W2 + b2
  k_gemm_res<4096><<<dim3(BS_ / 32, E_ / 64), 32, 0, stream>>>(fb, W2t, x1, b2, out);
}
